// LORICausalSelfAttention_75419625718333
// MI455X (gfx1250) — matrix operand order repaired and checked by reading, not compiled, not run on hardware
//
#include <hip/hip_runtime.h>

typedef __bf16 bf16_t;
typedef __attribute__((ext_vector_type(16))) __bf16 v16bf;
typedef __attribute__((ext_vector_type(8)))  __bf16 v8bf;
typedef __attribute__((ext_vector_type(8)))  float  v8f;

// Problem constants (from reference)
constexpr int Bn = 4, Tn = 2048, Cn = 2048;
constexpr int BT = Bn * Tn;            // 8192 rows
constexpr int NB = 16, BS = 128, Rr = 128;
constexpr int Hh = 16, HD = 128;

// ---------------------------------------------------------------------------
// WMMA helpers (CDNA5 v_wmma_f32_16x16x32_bf16, wave32)
// ---------------------------------------------------------------------------
__device__ __forceinline__ v8f wmma_acc(v16bf a, v16bf b, v8f c) {
  return __builtin_amdgcn_wmma_f32_16x16x32_bf16(
      /*neg_a=*/false, a, /*neg_b=*/false, b,
      /*c_mod=*/(short)0, c, /*reuse_a=*/false, /*reuse_b=*/false);
}

// A fragment: 16x32 bf16, row-major source, lda in elements.
__device__ __forceinline__ v16bf load_frag_a(const bf16_t* A, int lda, int lane) {
  int row  = lane & 15;
  int half = lane >> 4;
  const bf16_t* p = A + (size_t)row * lda + half * 8;
  v8bf lo = *(const v8bf*)(p);
  v8bf hi = *(const v8bf*)(p + 16);
  return __builtin_shufflevector(lo, hi, 0,1,2,3,4,5,6,7,8,9,10,11,12,13,14,15);
}

// B fragment: 32x16 bf16 from a transposed operand Bt[n][k] (ldb = K stride).
__device__ __forceinline__ v16bf load_frag_b(const bf16_t* Bt, int ldb, int lane) {
  int col  = lane & 15;
  int half = lane >> 4;
  const bf16_t* p = Bt + (size_t)col * ldb + half * 8;
  v8bf lo = *(const v8bf*)(p);
  v8bf hi = *(const v8bf*)(p + 16);
  return __builtin_shufflevector(lo, hi, 0,1,2,3,4,5,6,7,8,9,10,11,12,13,14,15);
}

// C/D fragment: lane l holds col N=l&15, rows M=(l>>4)*8 + r.
__device__ __forceinline__ void store_frag_f32(float* Cm, int ldc, int lane, v8f c) {
  int col = lane & 15, rb = (lane >> 4) * 8;
#pragma unroll
  for (int r = 0; r < 8; ++r) Cm[(size_t)(rb + r) * ldc + col] = c[r];
}

__device__ __forceinline__ void store_frag_bf16(bf16_t* Cm, int ldc, int lane, v8f c) {
  int col = lane & 15, rb = (lane >> 4) * 8;
#pragma unroll
  for (int r = 0; r < 8; ++r) Cm[(size_t)(rb + r) * ldc + col] = (bf16_t)c[r];
}

// Wave-internal LDS handoff (cross-lane data through LDS).
__device__ __forceinline__ void wave_lds_sync() {
  __builtin_amdgcn_wave_barrier();
  asm volatile("s_wait_dscnt 0" ::: "memory");
  __builtin_amdgcn_wave_barrier();
}

// ---------------------------------------------------------------------------
// Conversion kernels
// ---------------------------------------------------------------------------
__global__ void cvt_bf16_kernel(const float* __restrict__ in,
                                bf16_t* __restrict__ out, int n) {
  for (int i = blockIdx.x * blockDim.x + threadIdx.x; i < n;
       i += gridDim.x * blockDim.x)
    out[i] = (bf16_t)in[i];
}

// diag [4][NB][d][j] (f32) -> wdiagT [4][NB][j][d] (bf16)
__global__ void cvt_diagT_kernel(const float* __restrict__ in,
                                 bf16_t* __restrict__ out) {
  int i = blockIdx.x * blockDim.x + threadIdx.x;
  const int n = 4 * NB * BS * BS;
  if (i >= n) return;
  int j   = i & 127;
  int d   = (i >> 7) & 127;
  int blk = i >> 14;
  out[((size_t)blk << 14) + ((size_t)j << 7) + d] = (bf16_t)in[i];
}

// ---------------------------------------------------------------------------
// Low-rank "left" GEMM: U[M,R] = X[M,C] * L[R,C]^T
// 64x32 tile per wave (acc = 8 frags = 64 VGPRs; no spills).
// ---------------------------------------------------------------------------
__global__ void lr_u_kernel(const bf16_t* __restrict__ X,
                            const bf16_t* __restrict__ L,
                            bf16_t* __restrict__ U) {
  const int lane = threadIdx.x & 31;
  const int wave = threadIdx.x >> 5;
  const int TN = Rr / 32;                       // 4
  const int TOTAL = (BT / 64) * TN;             // 512
  int tile = blockIdx.x * 8 + wave;
  if (tile >= TOTAL) return;
  int m0 = (tile / TN) * 64;
  int n0 = (tile % TN) * 32;

  v8f acc[4][2] = {};
#pragma unroll 1
  for (int k0 = 0; k0 < Cn; k0 += 32) {
    __builtin_prefetch(X + (size_t)m0 * Cn + k0 + 256, 0, 0);
    v16bf a[4];
#pragma unroll
    for (int i = 0; i < 4; ++i)
      a[i] = load_frag_a(X + (size_t)(m0 + 16 * i) * Cn + k0, Cn, lane);
#pragma unroll
    for (int j = 0; j < 2; ++j) {
      v16bf b = load_frag_b(L + (size_t)(n0 + 16 * j) * Cn + k0, Cn, lane);
#pragma unroll
      for (int i = 0; i < 4; ++i) acc[i][j] = wmma_acc(a[i], b, acc[i][j]);
    }
  }
#pragma unroll
  for (int i = 0; i < 4; ++i)
#pragma unroll
    for (int j = 0; j < 2; ++j)
      store_frag_bf16(U + (size_t)(m0 + 16 * i) * Rr + n0 + 16 * j, Rr, lane,
                      acc[i][j]);
}

// ---------------------------------------------------------------------------
// Fused lori_fc output: block-diag GEMM (K=128) + low-rank right GEMM (K=128)
// + right_b + bias_p. 64x32 tile per wave.
// ---------------------------------------------------------------------------
template <bool OUT_F32>
__global__ void fc_out_kernel(const bf16_t* __restrict__ X,
                              const bf16_t* __restrict__ DT,   // [NB][128][128]
                              const bf16_t* __restrict__ U,    // [M,R]
                              const bf16_t* __restrict__ RT,   // [C][R]
                              const float* __restrict__ rb,
                              const float* __restrict__ bp,
                              bf16_t* __restrict__ outb,
                              float* __restrict__ outf) {
  const int lane = threadIdx.x & 31;
  const int wave = threadIdx.x >> 5;
  const int TN = Cn / 32;                       // 64
  const int TOTAL = (BT / 64) * TN;             // 8192
  int tile = blockIdx.x * 8 + wave;
  if (tile >= TOTAL) return;
  int m0 = (tile / TN) * 64;
  int n0 = (tile % TN) * 32;
  int blk = n0 >> 7;
  int jj0 = n0 & 127;

  v8f acc[4][2] = {};

  // GEMM1: block-diagonal path, K = 128
  const bf16_t* Xb = X + (size_t)m0 * Cn + blk * 128;
  const bf16_t* Dt = DT + (size_t)blk * BS * BS;
#pragma unroll 1
  for (int k0 = 0; k0 < BS; k0 += 32) {
    v16bf a[4];
#pragma unroll
    for (int i = 0; i < 4; ++i)
      a[i] = load_frag_a(Xb + (size_t)(16 * i) * Cn + k0, Cn, lane);
#pragma unroll
    for (int j = 0; j < 2; ++j) {
      v16bf b = load_frag_b(Dt + (size_t)(jj0 + 16 * j) * BS + k0, BS, lane);
#pragma unroll
      for (int i = 0; i < 4; ++i) acc[i][j] = wmma_acc(a[i], b, acc[i][j]);
    }
  }

  // GEMM2: low-rank right path, K = 128
#pragma unroll 1
  for (int k0 = 0; k0 < Rr; k0 += 32) {
    v16bf a[4];
#pragma unroll
    for (int i = 0; i < 4; ++i)
      a[i] = load_frag_a(U + (size_t)(m0 + 16 * i) * Rr + k0, Rr, lane);
#pragma unroll
    for (int j = 0; j < 2; ++j) {
      v16bf b = load_frag_b(RT + (size_t)(n0 + 16 * j) * Rr + k0, Rr, lane);
#pragma unroll
      for (int i = 0; i < 4; ++i) acc[i][j] = wmma_acc(a[i], b, acc[i][j]);
    }
  }

  // Biases (per output column)
#pragma unroll
  for (int j = 0; j < 2; ++j) {
    int col = n0 + 16 * j + (lane & 15);
    float bsum = rb[col] + bp[col];
#pragma unroll
    for (int i = 0; i < 4; ++i)
#pragma unroll
      for (int r = 0; r < 8; ++r) acc[i][j][r] += bsum;
  }

#pragma unroll
  for (int i = 0; i < 4; ++i)
#pragma unroll
    for (int j = 0; j < 2; ++j) {
      if (OUT_F32)
        store_frag_f32(outf + (size_t)(m0 + 16 * i) * Cn + n0 + 16 * j, Cn,
                       lane, acc[i][j]);
      else
        store_frag_bf16(outb + (size_t)(m0 + 16 * i) * Cn + n0 + 16 * j, Cn,
                        lane, acc[i][j]);
    }
}

// ---------------------------------------------------------------------------
// Flash attention (causal). One wave per 32-row Q tile of one (b,h).
// Per-wave LDS: Qs (32x128 bf16) | S (32x64 f32) | P (32x64 bf16)
//             | Vt (128x64 bf16) | rowscale (32 f32)
// ---------------------------------------------------------------------------
constexpr int ATT_WAVES = 4;
constexpr int WAVE_SMEM =
    (32 * 128 * 2) + (32 * 64 * 4) + (32 * 64 * 2) + (128 * 64 * 2) + 32 * 4 + 128;
// = 8192 + 8192 + 4096 + 16384 + 128 (+pad) = 37120

__global__ void attn_kernel(const bf16_t* __restrict__ Q,
                            const bf16_t* __restrict__ K,
                            const bf16_t* __restrict__ V,
                            bf16_t* __restrict__ Y) {
  extern __shared__ char smem[];
  const int lane = threadIdx.x & 31;
  const int wave = threadIdx.x >> 5;
  int gw = blockIdx.x * ATT_WAVES + wave;
  const int QT = Tn / 32;                                   // 64
  if (gw >= Bn * Hh * QT) return;
  int qt = gw & (QT - 1);
  int bh = gw / QT;
  int h = bh & (Hh - 1);
  int b = bh / Hh;

  char* wbase = smem + (size_t)wave * WAVE_SMEM;
  bf16_t* Qs       = (bf16_t*)wbase;                 // [32][128]
  float*  S        = (float*)(Qs + 32 * 128);        // [32][64]
  bf16_t* P        = (bf16_t*)(S + 32 * 64);         // [32][64]
  bf16_t* Vt       = P + 32 * 64;                    // [128][64]
  float*  rowscale = (float*)(Vt + 128 * 64);        // [32]

  const bf16_t* Qp = Q + (size_t)b * Tn * Cn + (size_t)h * HD;
  const bf16_t* Kp = K + (size_t)b * Tn * Cn + (size_t)h * HD;
  const bf16_t* Vp = V + (size_t)b * Tn * Cn + (size_t)h * HD;
  int q0 = qt * 32;

  // Stage Q tile (32 x 128) into LDS once.
  for (int idx = lane; idx < 32 * 16; idx += 32) {
    int row = idx >> 4;
    int cg  = (idx & 15) * 8;
    *(v8bf*)(Qs + (size_t)row * 128 + cg) =
        *(const v8bf*)(Qp + (size_t)(q0 + row) * Cn + cg);
  }
  wave_lds_sync();

  v8f o[2][8] = {};
  const float NEG_INF = -__builtin_inff();
  float m_r = NEG_INF;
  float l_r = 0.f;
  const float softscale = 0.08838834764831845f;  // 1/sqrt(128)

  const int ktend = q0 / 64;  // last (possibly partial) 64-key tile
  for (int kt = 0; kt <= ktend; ++kt) {
    int k0 = kt * 64;

    // Stage V tile (64 x 128) transposed -> Vt[c][k]
    for (int idx = lane; idx < 64 * 16; idx += 32) {
      int row = idx >> 4;
      int cg  = (idx & 15) * 8;
      v8bf vv = *(const v8bf*)(Vp + (size_t)(k0 + row) * Cn + cg);
#pragma unroll
      for (int e = 0; e < 8; ++e) Vt[(size_t)(cg + e) * 64 + row] = vv[e];
    }

    // S = Qtile * Ktile^T  (32 x 64), spill each fragment to LDS as produced.
#pragma unroll
    for (int fn = 0; fn < 4; ++fn) {
      v16bf bK[4];
#pragma unroll
      for (int ks = 0; ks < 4; ++ks)
        bK[ks] = load_frag_b(Kp + (size_t)(k0 + 16 * fn) * Cn + ks * 32, Cn,
                             lane);
#pragma unroll
      for (int fm = 0; fm < 2; ++fm) {
        v8f s = {};
#pragma unroll
        for (int ks = 0; ks < 4; ++ks) {
          v16bf qa = load_frag_a(Qs + (size_t)(16 * fm) * 128 + ks * 32, 128,
                                 lane);
          s = wmma_acc(qa, bK[ks], s);
        }
        int col = lane & 15, rb2 = (lane >> 4) * 8;
        float* Sp = S + (size_t)(fm * 16 + rb2) * 64 + fn * 16 + col;
#pragma unroll
        for (int r = 0; r < 8; ++r) Sp[(size_t)r * 64] = s[r] * softscale;
      }
    }
    wave_lds_sync();

    // Online softmax: lane owns row = lane. Vectorized LDS access (v8f / v8bf).
    {
      int row = lane;
      const float* Srow = S + (size_t)row * 64;
      bf16_t* Prow = P + (size_t)row * 64;
      float mx = m_r;
      float sum = 0.f;
      if (kt < ktend) {
        // Full (unmasked) tile.
#pragma unroll
        for (int c8 = 0; c8 < 8; ++c8) {
          v8f sv = *(const v8f*)(Srow + c8 * 8);
#pragma unroll
          for (int e = 0; e < 8; ++e) mx = (sv[e] > mx) ? sv[e] : mx;
        }
        float scale = __expf(m_r - mx);
#pragma unroll
        for (int c8 = 0; c8 < 8; ++c8) {
          v8f sv = *(const v8f*)(Srow + c8 * 8);
          v8bf pv;
#pragma unroll
          for (int e = 0; e < 8; ++e) {
            float p = __expf(sv[e] - mx);
            sum += p;
            pv[e] = (bf16_t)p;
          }
          *(v8bf*)(Prow + c8 * 8) = pv;
        }
        l_r = l_r * scale + sum;
        m_r = mx;
        rowscale[row] = scale;
      } else {
        // Diagonal tile: mask columns >= kmax.
        int kmax = (q0 + row) - k0 + 1;  // in [1, 64]
#pragma unroll
        for (int c8 = 0; c8 < 8; ++c8) {
          v8f sv = *(const v8f*)(Srow + c8 * 8);
#pragma unroll
          for (int e = 0; e < 8; ++e) {
            float v = (c8 * 8 + e < kmax) ? sv[e] : NEG_INF;
            mx = (v > mx) ? v : mx;
          }
        }
        float scale = __expf(m_r - mx);
#pragma unroll
        for (int c8 = 0; c8 < 8; ++c8) {
          v8f sv = *(const v8f*)(Srow + c8 * 8);
          v8bf pv;
#pragma unroll
          for (int e = 0; e < 8; ++e) {
            float p = (c8 * 8 + e < kmax) ? __expf(sv[e] - mx) : 0.f;
            sum += p;
            pv[e] = (bf16_t)p;
          }
          *(v8bf*)(Prow + c8 * 8) = pv;
        }
        l_r = l_r * scale + sum;
        m_r = mx;
        rowscale[row] = scale;
      }
    }
    wave_lds_sync();

    // Rescale O accumulators by per-row scale.
    {
      int rb2 = (lane >> 4) * 8;
#pragma unroll
      for (int fm = 0; fm < 2; ++fm)
#pragma unroll
        for (int r = 0; r < 8; ++r) {
          float rs = rowscale[fm * 16 + rb2 + r];
#pragma unroll
          for (int fn = 0; fn < 8; ++fn) o[fm][fn][r] *= rs;
        }
    }

    // O += P * V   (M=32, K=64, N=128) from LDS-staged P (A) and Vt (B^T).
#pragma unroll
    for (int ks = 0; ks < 2; ++ks) {
      v16bf aP[2];
#pragma unroll
      for (int fm = 0; fm < 2; ++fm)
        aP[fm] = load_frag_a(P + (size_t)(fm * 16) * 64 + ks * 32, 64, lane);
#pragma unroll
      for (int fn = 0; fn < 8; ++fn) {
        v16bf bV = load_frag_b(Vt + (size_t)(fn * 16) * 64 + ks * 32, 64, lane);
#pragma unroll
        for (int fm = 0; fm < 2; ++fm)
          o[fm][fn] = wmma_acc(aP[fm], bV, o[fm][fn]);
      }
    }
  }

  // Final normalization 1/l and store.
  rowscale[lane] = 1.0f / l_r;
  wave_lds_sync();
  {
    int col = lane & 15, rb2 = (lane >> 4) * 8;
#pragma unroll
    for (int fm = 0; fm < 2; ++fm)
#pragma unroll
      for (int r = 0; r < 8; ++r) {
        float inv = rowscale[fm * 16 + rb2 + r];
        int row = q0 + fm * 16 + rb2 + r;
        bf16_t* yp = Y + (size_t)b * Tn * Cn + (size_t)row * Cn + h * HD;
#pragma unroll
        for (int fn = 0; fn < 8; ++fn)
          yp[fn * 16 + col] = (bf16_t)(o[fm][fn][r] * inv);
      }
  }
}

// ---------------------------------------------------------------------------
// Host launcher
// ---------------------------------------------------------------------------
extern "C" void kernel_launch(void* const* d_in, const int* in_sizes, int n_in,
                              void* d_out, int out_size, void* d_ws,
                              size_t ws_size, hipStream_t stream) {
  const float* x     = (const float*)d_in[0];
  const float* diag  = (const float*)d_in[1];
  const float* leftw = (const float*)d_in[2];
  const float* rightw= (const float*)d_in[3];
  const float* rbias = (const float*)d_in[4];
  const float* pbias = (const float*)d_in[5];
  (void)in_sizes; (void)n_in; (void)out_size; (void)ws_size;

  char* ws = (char*)d_ws;
  size_t off = 0;
  auto alloc = [&](size_t bytes) -> void* {
    void* p = ws + off;
    off += (bytes + 255) & ~(size_t)255;
    return p;
  };

  bf16_t* xb  = (bf16_t*)alloc((size_t)BT * Cn * 2);
  bf16_t* qb  = (bf16_t*)alloc((size_t)BT * Cn * 2);
  bf16_t* kb  = (bf16_t*)alloc((size_t)BT * Cn * 2);
  bf16_t* vb  = (bf16_t*)alloc((size_t)BT * Cn * 2);
  bf16_t* yb  = (bf16_t*)alloc((size_t)BT * Cn * 2);
  bf16_t* ub  = (bf16_t*)alloc((size_t)BT * Rr * 2);
  bf16_t* wdT = (bf16_t*)alloc((size_t)4 * NB * BS * BS * 2);
  bf16_t* wL  = (bf16_t*)alloc((size_t)4 * Rr * Cn * 2);
  bf16_t* wR  = (bf16_t*)alloc((size_t)4 * Cn * Rr * 2);

  // 1) conversions
  {
    int n = BT * Cn;
    cvt_bf16_kernel<<<8192, 256, 0, stream>>>(x, xb, n);
    int nw = 4 * Rr * Cn;
    cvt_bf16_kernel<<<2048, 256, 0, stream>>>(leftw, wL, nw);
    cvt_bf16_kernel<<<2048, 256, 0, stream>>>(rightw, wR, nw);
    int nd = 4 * NB * BS * BS;
    cvt_diagT_kernel<<<(nd + 255) / 256, 256, 0, stream>>>(diag, wdT);
  }

  const int U_BLOCKS  = ((BT / 64) * (Rr / 32) + 7) / 8;   // 64
  const int FC_BLOCKS = ((BT / 64) * (Cn / 32) + 7) / 8;   // 1024

  // 2) q, k, v projections
  bf16_t* qkv[3] = {qb, kb, vb};
  for (int i = 0; i < 3; ++i) {
    lr_u_kernel<<<U_BLOCKS, 256, 0, stream>>>(xb, wL + (size_t)i * Rr * Cn, ub);
    fc_out_kernel<false><<<FC_BLOCKS, 256, 0, stream>>>(
        xb, wdT + (size_t)i * NB * BS * BS, ub, wR + (size_t)i * Cn * Rr,
        rbias + (size_t)i * Cn, pbias + (size_t)i * Cn, qkv[i], nullptr);
  }

  // 3) causal attention -> yb
  {
    const int ATT_TILES = Bn * Hh * (Tn / 32);             // 4096
    const int blocks = ATT_TILES / ATT_WAVES;              // 1024
    attn_kernel<<<blocks, ATT_WAVES * 32, ATT_WAVES * WAVE_SMEM, stream>>>(
        qb, kb, vb, yb);
  }

  // 4) output projection (f32 result)
  lr_u_kernel<<<U_BLOCKS, 256, 0, stream>>>(yb, wL + (size_t)3 * Rr * Cn, ub);
  fc_out_kernel<true><<<FC_BLOCKS, 256, 0, stream>>>(
      yb, wdT + (size_t)3 * NB * BS * BS, ub, wR + (size_t)3 * Cn * Rr,
      rbias + (size_t)3 * Cn, pbias + (size_t)3 * Cn, nullptr, (float*)d_out);
}
